// LSTMAE_87179246174768
// MI455X (gfx1250) — matrix operand order repaired and checked by reading, not compiled, not run on hardware
//
#include <hip/hip_runtime.h>

typedef _Float16 v8h  __attribute__((ext_vector_type(8)));
typedef _Float16 v16h __attribute__((ext_vector_type(16)));
typedef float    v8f  __attribute__((ext_vector_type(8)));

#define TSEQ 4096
#define HIDN 32
#define ZDIM 16
#define NBAT 256

// Fast activations: avoid IEEE division expansions (v_div_scale/v_div_fmas).
// Prefer gfx1250's native v_tanh_f32 if the toolchain exposes it; otherwise a
// single v_rcp_f32 via __builtin_amdgcn_rcpf.
__device__ __forceinline__ float fast_rcp(float x) {
    return __builtin_amdgcn_rcpf(x);
}

#if __has_builtin(__builtin_amdgcn_tanhf)
__device__ __forceinline__ float ftanh(float x) {
    return __builtin_amdgcn_tanhf(x);
}
__device__ __forceinline__ float fsigmoid(float x) {
    return fmaf(0.5f, __builtin_amdgcn_tanhf(0.5f * x), 0.5f);
}
#else
__device__ __forceinline__ float ftanh(float x) {
    // tanh(x) = 2/(1+exp(-2x)) - 1
    float e = __expf(-2.0f * x);
    return fmaf(2.0f, fast_rcp(1.0f + e), -1.0f);
}
__device__ __forceinline__ float fsigmoid(float x) {
    return fast_rcp(1.0f + __expf(-x));
}
#endif

// grid = 16 blocks (one 16-row batch tile each), block = 64 threads (2 waves).
// Wave w owns hidden columns [16w, 16w+16): computes gate tiles {w, w+2, w+4, w+6}
// (= i,f,g,o for those columns) so the c/h update is wave-local.
__global__ __launch_bounds__(64, 1)
void lstm_ae_kernel(const float* __restrict__ x,
                    const float* __restrict__ eWih, const float* __restrict__ eWhh,
                    const float* __restrict__ ebih, const float* __restrict__ ebhh,
                    const float* __restrict__ tozW, const float* __restrict__ tozb,
                    const float* __restrict__ fzW,  const float* __restrict__ fzb,
                    const float* __restrict__ dWih, const float* __restrict__ dWhh,
                    const float* __restrict__ dbih, const float* __restrict__ dbhh,
                    float* __restrict__ out)
{
    // ping-pong h exchange buffer (f16), row stride 40 halves = 80B (16B aligned,
    // bank shift 20 per row). Plus f32 h_T staging for the bottleneck.
    __shared__ _Float16 hbuf[2][16][40];
    __shared__ float    hT[16][HIDN + 1];

    const int lane = threadIdx.x & 31;
    const int w    = threadIdx.x >> 5;      // wave id 0/1
    const int nlo  = lane & 15;
    const int hi   = (lane >> 4) & 1;       // lane half
    const int m0   = blockIdx.x * 16;       // batch tile base row

    // ---- one-time: Whh^T tiles as f16 B operands; Wih/bias per gate column ----
    v16h Bt[4];
    float wih[4], bias[4];
    const int tiles[4] = { w, 2 + w, 4 + w, 6 + w };   // i, f, g, o
    #pragma unroll
    for (int q = 0; q < 4; ++q) {
        const int grow = tiles[q] * 16 + nlo;          // gate (column) index = B's N
        const int kb   = hi * 16;                       // lanes 0-15: K 0..15, 16-31: K 16..31
        v16h bv;
        #pragma unroll
        for (int j = 0; j < 16; ++j)
            bv[j] = (_Float16)eWhh[grow * HIDN + kb + j];   // B[k][n] = Whh[n_gate][k]
        Bt[q]   = bv;
        wih[q]  = eWih[grow];                   // IN_DIM == 1
        bias[q] = ebih[grow] + ebhh[grow];
    }

    // bpermute byte-indices: C-layout element (VGPR r, lane) is row M = r + 8*hi
    int xidx[8];
    #pragma unroll
    for (int r = 0; r < 8; ++r) xidx[r] = (r + 8 * hi) * 4;

    // state: A = h_{t-1} (f16 A-layout), cst = c for this wave's 16 columns
    v16h A;
    #pragma unroll
    for (int j = 0; j < 16; ++j) A[j] = (_Float16)0.0f;
    v8f cst = {};
    float hcur[8];
    #pragma unroll
    for (int r = 0; r < 8; ++r) hcur[r] = 0.0f;

    const float* xrow = x + (size_t)(m0 + nlo) * TSEQ;   // this lane's batch row of x

    for (int t = 0; t < TSEQ; ++t) {
        if ((t & 63) == 0) __builtin_prefetch(xrow + t + 64, 0, 0);

        // x[b][t] for my row, then broadcast rows 0..7 / 8..15 across lanes
        const int xb = __float_as_int(xrow[t]);
        float xr[8];
        #pragma unroll
        for (int r = 0; r < 8; ++r)
            xr[r] = __int_as_float(__builtin_amdgcn_ds_bpermute(xidx[r], xb));

        // gates = (x*Wih + b) + h_{t-1} @ Whh^T : fold x-term into WMMA's C input
        v8f g[4];
        #pragma unroll
        for (int q = 0; q < 4; ++q) {
            v8f cinit;
            #pragma unroll
            for (int r = 0; r < 8; ++r)
                cinit[r] = fmaf(xr[r], wih[q], bias[q]);
            g[q] = __builtin_amdgcn_wmma_f32_16x16x32_f16(
                       false, A, false, Bt[q], (short)0, cinit, false, false);
        }

        // activations + state update (all wave-local, f32)
        #pragma unroll
        for (int r = 0; r < 8; ++r) {
            const float iv = fsigmoid(g[0][r]);
            const float fv = fsigmoid(g[1][r]);
            const float gv = ftanh   (g[2][r]);
            const float ov = fsigmoid(g[3][r]);
            const float cv = fv * cst[r] + iv * gv;
            cst[r]  = cv;
            hcur[r] = ov * ftanh(cv);
        }

        // exchange h through LDS (ping-pong) and rebuild the f16 A operand
        const int buf = t & 1;
        #pragma unroll
        for (int r = 0; r < 8; ++r)
            hbuf[buf][r + 8 * hi][w * 16 + nlo] = (_Float16)hcur[r];
        __syncthreads();
        {
            const _Float16* rowp = &hbuf[buf][nlo][0];   // A row M = lane&15
            const int k0 = hi * 16;                       // lanes0-15: K0..15, 16-31: K16..31
            v8h lo = *(const v8h*)(rowp + k0);
            v8h hh = *(const v8h*)(rowp + k0 + 8);
            A = __builtin_shufflevector(lo, hh, 0,1,2,3,4,5,6,7,8,9,10,11,12,13,14,15);
        }
    }

    // ---- stage h_T (f32) for the bottleneck ----
    #pragma unroll
    for (int r = 0; r < 8; ++r)
        hT[r + 8 * hi][w * 16 + nlo] = hcur[r];
    __syncthreads();

    // ---- bottleneck + decoder: lanes 0..15 of wave 0, one batch row each ----
    if (threadIdx.x < 16) {
        const int b = threadIdx.x;
        float hrow[HIDN];
        #pragma unroll
        for (int k = 0; k < HIDN; ++k) hrow[k] = hT[b][k];

        float z[ZDIM];
        #pragma unroll
        for (int j = 0; j < ZDIM; ++j) {
            float acc = tozb[j];
            #pragma unroll
            for (int k = 0; k < HIDN; ++k) acc = fmaf(hrow[k], tozW[j * HIDN + k], acc);
            z[j] = acc;
        }
        float h0v[HIDN];
        #pragma unroll
        for (int k = 0; k < HIDN; ++k) {
            float acc = fzb[k];
            #pragma unroll
            for (int j = 0; j < ZDIM; ++j) acc = fmaf(z[j], fzW[k * ZDIM + j], acc);
            h0v[k] = ftanh(acc);
        }
        // decoder xproj is constant over t (input is broadcast h0); hidden dim = 1
        float xp[4], whd[4];
        #pragma unroll
        for (int q = 0; q < 4; ++q) {
            float acc = dbih[q] + dbhh[q];
            #pragma unroll
            for (int k = 0; k < HIDN; ++k) acc = fmaf(h0v[k], dWih[q * HIDN + k], acc);
            xp[q]  = acc;
            whd[q] = dWhh[q];
        }

        float hh = 0.0f, cc = 0.0f;
        float* op = out + (size_t)(m0 + b) * TSEQ;
        for (int t = 0; t < TSEQ; ++t) {
            const float iv = fsigmoid(fmaf(hh, whd[0], xp[0]));
            const float fv = fsigmoid(fmaf(hh, whd[1], xp[1]));
            const float gv = ftanh   (fmaf(hh, whd[2], xp[2]));
            const float ov = fsigmoid(fmaf(hh, whd[3], xp[3]));
            cc = fv * cc + iv * gv;
            hh = ov * ftanh(cc);
            op[t] = hh;
        }
    }
}

extern "C" void kernel_launch(void* const* d_in, const int* in_sizes, int n_in,
                              void* d_out, int out_size, void* d_ws, size_t ws_size,
                              hipStream_t stream) {
    (void)in_sizes; (void)n_in; (void)out_size; (void)d_ws; (void)ws_size;
    const float* x    = (const float*)d_in[0];
    const float* eWih = (const float*)d_in[1];
    const float* eWhh = (const float*)d_in[2];
    const float* ebih = (const float*)d_in[3];
    const float* ebhh = (const float*)d_in[4];
    const float* tozW = (const float*)d_in[5];
    const float* tozb = (const float*)d_in[6];
    const float* fzW  = (const float*)d_in[7];
    const float* fzb  = (const float*)d_in[8];
    const float* dWih = (const float*)d_in[9];
    const float* dWhh = (const float*)d_in[10];
    const float* dbih = (const float*)d_in[11];
    const float* dbhh = (const float*)d_in[12];
    float* out = (float*)d_out;

    dim3 grid(NBAT / 16), block(64);
    lstm_ae_kernel<<<grid, block, 0, stream>>>(x, eWih, eWhh, ebih, ebhh,
                                               tozW, tozb, fzW, fzb,
                                               dWih, dWhh, dbih, dbhh, out);
}
